// PathwayGAT2_38465727103847
// MI455X (gfx1250) — matrix operand order repaired and checked by reading, not compiled, not run on hardware
//
#include <hip/hip_runtime.h>
#include <hip/hip_bf16.h>

#define N_NODES 10000
#define E_EDGES 320000
#define ETOT    (E_EDGES + N_NODES)
#define F_IN    256
#define HID     128
#define NEG_SLOPE 0.2f

typedef __attribute__((ext_vector_type(16))) __bf16 v16bf;
typedef __attribute__((ext_vector_type(8)))  float  v8f;

struct bvec32 { uint4 lo; uint4 hi; };   // 32 bytes == v16bf

// ---------------------------------------------------------------------------
// Prep: fp32 -> bf16 copy, and fp32 [K,C] -> bf16 [C,K] transpose.
// ---------------------------------------------------------------------------
__global__ void pg_f32_to_bf16(const float* __restrict__ src,
                               __hip_bfloat16* __restrict__ dst, int n) {
    int i = blockIdx.x * blockDim.x + threadIdx.x;
    if (i < n) dst[i] = __float2bfloat16(src[i]);
}

__global__ void pg_transpose_bf16(const float* __restrict__ W,
                                  __hip_bfloat16* __restrict__ Wt,
                                  int K, int C) {
    int i = blockIdx.x * blockDim.x + threadIdx.x;
    if (i < K * C) {
        int k = i / C, c = i % C;
        Wt[(size_t)c * K + k] = __float2bfloat16(W[i]);
    }
}

// ---------------------------------------------------------------------------
// GEMM: C[M,128] = A[M,K](bf16) * B[K,128], B given as Bt[128,K] (bf16).
// One block (8 waves) per 16-row stripe; wave w owns columns [16w,16w+16).
// A fragment (16x32 bf16): lane m=lane&15 is row; elems 0..7 = K(half*8..),
// elems 8..15 = K(16+half*8..)  -> two contiguous 16B loads.
// B fragment (32x16 bf16): lane n=lane&15 is col; same K order as A
// -> two contiguous 16B loads from the transposed weight row.
// ---------------------------------------------------------------------------
template <int K>
__global__ void pg_gemm_wmma_bf16(const __hip_bfloat16* __restrict__ A,
                                  const __hip_bfloat16* __restrict__ Bt,
                                  float* __restrict__ C) {
    const int lane = threadIdx.x & 31;
    const int wave = threadIdx.x >> 5;      // 0..7 -> column tile
    const int m    = lane & 15;
    const int half = lane >> 4;
    const int row0 = blockIdx.x * 16;
    const int col  = wave * 16 + m;

    const __hip_bfloat16* arow = A  + (size_t)(row0 + m) * K;
    const __hip_bfloat16* brow = Bt + (size_t)col * K;

    v8f acc = {};
#pragma unroll
    for (int k0 = 0; k0 < K; k0 += 32) {
        bvec32 ab, bb;
        ab.lo = *(const uint4*)(arow + k0 + half * 8);
        ab.hi = *(const uint4*)(arow + k0 + 16 + half * 8);
        bb.lo = *(const uint4*)(brow + k0 + half * 8);
        bb.hi = *(const uint4*)(brow + k0 + 16 + half * 8);
        v16bf a = __builtin_bit_cast(v16bf, ab);
        v16bf b = __builtin_bit_cast(v16bf, bb);
        acc = __builtin_amdgcn_wmma_f32_16x16x32_bf16(
            false, a, false, b, (short)0, acc, false, false);
    }

    // D layout: VGPR v -> row (8*half + v), column = col.
    float* crow = C + (size_t)(row0 + half * 8) * HID + col;
#pragma unroll
    for (int v = 0; v < 8; ++v) crow[(size_t)v * HID] = acc[v];
}

// ---------------------------------------------------------------------------
// Per-node attention scores: a_s[n] = h[n]·att_src, a_d[n] = h[n]·att_dst.
// One wave per node, float4 per lane, shuffle reduction.
// ---------------------------------------------------------------------------
__global__ void pg_att_dots(const float* __restrict__ h,
                            const float* __restrict__ att_s,
                            const float* __restrict__ att_d,
                            float* __restrict__ as, float* __restrict__ ad) {
    int wid  = (blockIdx.x * blockDim.x + threadIdx.x) >> 5;
    int lane = threadIdx.x & 31;
    if (wid >= N_NODES) return;
    float4 hv = *(const float4*)(h + (size_t)wid * HID + lane * 4);
    float4 sv = *(const float4*)(att_s + lane * 4);
    float4 dv = *(const float4*)(att_d + lane * 4);
    float ss = hv.x * sv.x + hv.y * sv.y + hv.z * sv.z + hv.w * sv.w;
    float dd = hv.x * dv.x + hv.y * dv.y + hv.z * dv.z + hv.w * dv.w;
#pragma unroll
    for (int off = 16; off; off >>= 1) {
        ss += __shfl_down(ss, off, 32);
        dd += __shfl_down(dd, off, 32);
    }
    if (lane == 0) { as[wid] = ss; ad[wid] = dd; }
}

// ---------------------------------------------------------------------------
// Segment softmax + aggregation (3 passes over edges, self loops appended).
// ---------------------------------------------------------------------------
__device__ __forceinline__ void pg_edge_sd(const int* __restrict__ ei, int e,
                                           int& s, int& d) {
    if (e < E_EDGES) { s = ei[e]; d = ei[E_EDGES + e]; }
    else             { s = e - E_EDGES; d = s; }
}

__device__ __forceinline__ void pg_atomic_max_f32(float* addr, float val) {
    unsigned int* ua  = (unsigned int*)addr;
    unsigned int  old = *((volatile unsigned int*)ua);
    while (__uint_as_float(old) < val) {
        unsigned int assumed = old;
        old = atomicCAS(ua, assumed, __float_as_uint(val));
        if (old == assumed) break;
    }
}

__global__ void pg_init_seg(float* __restrict__ agg, float* __restrict__ mbuf,
                            float* __restrict__ zbuf) {
    int i = blockIdx.x * blockDim.x + threadIdx.x;
    if (i < N_NODES * HID) agg[i] = 0.0f;
    if (i < N_NODES) { mbuf[i] = -__builtin_huge_valf(); zbuf[i] = 0.0f; }
}

__global__ void pg_edge_max(const int* __restrict__ ei,
                            const float* __restrict__ as,
                            const float* __restrict__ ad,
                            float* __restrict__ ebuf, float* __restrict__ mbuf) {
    int e = blockIdx.x * blockDim.x + threadIdx.x;
    if (e >= ETOT) return;
    int s, d; pg_edge_sd(ei, e, s, d);
    float v = as[s] + ad[d];
    v = v > 0.0f ? v : NEG_SLOPE * v;     // leaky_relu
    ebuf[e] = v;
    pg_atomic_max_f32(&mbuf[d], v);
}

__global__ void pg_edge_exp(const int* __restrict__ ei,
                            float* __restrict__ ebuf,
                            const float* __restrict__ mbuf,
                            float* __restrict__ zbuf) {
    int e = blockIdx.x * blockDim.x + threadIdx.x;
    if (e >= ETOT) return;
    int s, d; pg_edge_sd(ei, e, s, d);
    float ex = __expf(ebuf[e] - mbuf[d]);
    ebuf[e] = ex;
    atomicAdd(&zbuf[d], ex);
}

__global__ void pg_edge_scatter(const int* __restrict__ ei,
                                const float* __restrict__ ebuf,
                                const float* __restrict__ zbuf,
                                const float* __restrict__ h,
                                float* __restrict__ agg) {
    int wid  = (blockIdx.x * blockDim.x + threadIdx.x) >> 5;
    int lane = threadIdx.x & 31;
    if (wid >= ETOT) return;
    int s, d; pg_edge_sd(ei, wid, s, d);
    float alpha = ebuf[wid] / zbuf[d];
    float4 hv = *(const float4*)(h + (size_t)s * HID + lane * 4);
    float* dst = agg + (size_t)d * HID + lane * 4;
    atomicAdd(dst + 0, alpha * hv.x);
    atomicAdd(dst + 1, alpha * hv.y);
    atomicAdd(dst + 2, alpha * hv.z);
    atomicAdd(dst + 3, alpha * hv.w);
}

// ---------------------------------------------------------------------------
// Epilogues.
// ---------------------------------------------------------------------------
__global__ void pg_relu_bias_bf16(const float* __restrict__ agg,
                                  const float* __restrict__ bias,
                                  __hip_bfloat16* __restrict__ dst, int n) {
    int i = blockIdx.x * blockDim.x + threadIdx.x;
    if (i < n)
        dst[i] = __float2bfloat16(fmaxf(agg[i] + bias[i % HID], 0.0f));
}

__global__ void pg_init_out(const float* __restrict__ bc, float* __restrict__ out) {
    if (blockIdx.x == 0 && threadIdx.x == 0) { out[0] = bc[0]; out[1] = bc[1]; }
}

// per node: s = relu(agg+b2)·Wr + br ; out[c] += s * Wc[n,c]
__global__ void pg_finalize(const float* __restrict__ agg,
                            const float* __restrict__ b2,
                            const float* __restrict__ Wr,
                            const float* __restrict__ br,
                            const float* __restrict__ Wc,
                            float* __restrict__ out) {
    int wid  = (blockIdx.x * blockDim.x + threadIdx.x) >> 5;
    int lane = threadIdx.x & 31;
    if (wid >= N_NODES) return;
    float4 av = *(const float4*)(agg + (size_t)wid * HID + lane * 4);
    float4 bv = *(const float4*)(b2 + lane * 4);
    float4 wv = *(const float4*)(Wr + lane * 4);
    float s = fmaxf(av.x + bv.x, 0.0f) * wv.x + fmaxf(av.y + bv.y, 0.0f) * wv.y +
              fmaxf(av.z + bv.z, 0.0f) * wv.z + fmaxf(av.w + bv.w, 0.0f) * wv.w;
#pragma unroll
    for (int off = 16; off; off >>= 1) s += __shfl_down(s, off, 32);
    if (lane == 0) {
        s += br[0];
        atomicAdd(&out[0], s * Wc[(size_t)wid * 2 + 0]);
        atomicAdd(&out[1], s * Wc[(size_t)wid * 2 + 1]);
    }
}

// ---------------------------------------------------------------------------
extern "C" void kernel_launch(void* const* d_in, const int* in_sizes, int n_in,
                              void* d_out, int out_size, void* d_ws, size_t ws_size,
                              hipStream_t stream) {
    const float* x        = (const float*)d_in[0];
    const int*   ei       = (const int*)  d_in[1];
    const float* W1       = (const float*)d_in[2];
    const float* att_src1 = (const float*)d_in[3];
    const float* att_dst1 = (const float*)d_in[4];
    const float* b1       = (const float*)d_in[5];
    const float* W2       = (const float*)d_in[6];
    const float* att_src2 = (const float*)d_in[7];
    const float* att_dst2 = (const float*)d_in[8];
    const float* b2       = (const float*)d_in[9];
    const float* Wr       = (const float*)d_in[10];
    const float* br       = (const float*)d_in[11];
    const float* Wc       = (const float*)d_in[12];
    const float* bc       = (const float*)d_in[13];
    float* out = (float*)d_out;

    char* w = (char*)d_ws;
    auto carve = [&](size_t bytes) -> char* {
        char* p = w;
        w += (bytes + 511) & ~(size_t)511;
        return p;
    };
    __hip_bfloat16* xb   = (__hip_bfloat16*)carve((size_t)N_NODES * F_IN * 2);
    __hip_bfloat16* W1t  = (__hip_bfloat16*)carve((size_t)HID * F_IN * 2);
    __hip_bfloat16* W2t  = (__hip_bfloat16*)carve((size_t)HID * HID * 2);
    float*          h    = (float*)carve((size_t)N_NODES * HID * 4);
    float*          agg  = (float*)carve((size_t)N_NODES * HID * 4);
    __hip_bfloat16* hb   = (__hip_bfloat16*)carve((size_t)N_NODES * HID * 2);
    float*          a_s  = (float*)carve((size_t)N_NODES * 4);
    float*          a_d  = (float*)carve((size_t)N_NODES * 4);
    float*          mbuf = (float*)carve((size_t)N_NODES * 4);
    float*          zbuf = (float*)carve((size_t)N_NODES * 4);
    float*          eb   = (float*)carve((size_t)ETOT * 4);

    const int TB = 256;
    dim3 blk(TB);
    int g_xconv = (N_NODES * F_IN + TB - 1) / TB;
    int g_w1    = (F_IN * HID + TB - 1) / TB;
    int g_w2    = (HID * HID + TB - 1) / TB;
    int g_nh    = (N_NODES * HID + TB - 1) / TB;
    int g_nodeW = (N_NODES + 7) / 8;       // one wave per node
    int g_edge  = (ETOT + TB - 1) / TB;
    int g_edgeW = (ETOT + 7) / 8;          // one wave per edge
    int g_gemm  = N_NODES / 16;            // 625, exact

    // ---- prep ----
    pg_f32_to_bf16   <<<g_xconv, blk, 0, stream>>>(x, xb, N_NODES * F_IN);
    pg_transpose_bf16<<<g_w1,    blk, 0, stream>>>(W1, W1t, F_IN, HID);
    pg_transpose_bf16<<<g_w2,    blk, 0, stream>>>(W2, W2t, HID, HID);

    // ---- layer 1 ----
    pg_gemm_wmma_bf16<F_IN><<<g_gemm, blk, 0, stream>>>(xb, W1t, h);
    pg_att_dots    <<<g_nodeW, blk, 0, stream>>>(h, att_src1, att_dst1, a_s, a_d);
    pg_init_seg    <<<g_nh,    blk, 0, stream>>>(agg, mbuf, zbuf);
    pg_edge_max    <<<g_edge,  blk, 0, stream>>>(ei, a_s, a_d, eb, mbuf);
    pg_edge_exp    <<<g_edge,  blk, 0, stream>>>(ei, eb, mbuf, zbuf);
    pg_edge_scatter<<<g_edgeW, blk, 0, stream>>>(ei, eb, zbuf, h, agg);
    pg_relu_bias_bf16<<<g_nh,  blk, 0, stream>>>(agg, b1, hb, N_NODES * HID);

    // ---- layer 2 ----
    pg_gemm_wmma_bf16<HID><<<g_gemm, blk, 0, stream>>>(hb, W2t, h);
    pg_att_dots    <<<g_nodeW, blk, 0, stream>>>(h, att_src2, att_dst2, a_s, a_d);
    pg_init_seg    <<<g_nh,    blk, 0, stream>>>(agg, mbuf, zbuf);
    pg_edge_max    <<<g_edge,  blk, 0, stream>>>(ei, a_s, a_d, eb, mbuf);
    pg_edge_exp    <<<g_edge,  blk, 0, stream>>>(ei, eb, mbuf, zbuf);
    pg_edge_scatter<<<g_edgeW, blk, 0, stream>>>(ei, eb, zbuf, h, agg);

    // ---- head ----
    pg_init_out<<<1, 32, 0, stream>>>(bc, out);
    pg_finalize<<<g_nodeW, blk, 0, stream>>>(agg, b2, Wr, br, Wc, out);
}
